// TransformerDiffusionBlock_8108898255593
// MI455X (gfx1250) — matrix operand order repaired and checked by reading, not compiled, not run on hardware
//
#include <hip/hip_runtime.h>

typedef unsigned short u16;
typedef __attribute__((ext_vector_type(16))) __bf16 v16bf;
typedef __attribute__((ext_vector_type(8)))  float  v8f;
typedef unsigned int v4u __attribute__((ext_vector_type(4)));
typedef int          v4i __attribute__((ext_vector_type(4)));
typedef int          v8i __attribute__((ext_vector_type(8)));

union FragBF { v16bf v; uint4 q[2]; };

static __device__ __forceinline__ float bf2f(u16 u) {
    union { unsigned int i; float f; } x; x.i = ((unsigned int)u) << 16; return x.f;
}
static __device__ __forceinline__ u16 f2bf(float f) {
    union { float f; unsigned int i; } x; x.f = f;
    unsigned int r = x.i + 0x7FFFu + ((x.i >> 16) & 1u);
    return (u16)(r >> 16);
}
static __device__ __forceinline__ float gelu_exact(float x) {
    return 0.5f * x * (1.0f + erff(x * 0.70710678118654752f));
}
static __device__ __forceinline__ v8f v8zero() {
    v8f z = {0.f,0.f,0.f,0.f,0.f,0.f,0.f,0.f}; return z;
}

// DPP16 ROW_XMASK butterfly steps (reduction within each 16-lane half)
template<int MASK>
static __device__ __forceinline__ float dpp_max(float v) {
    int t = __builtin_amdgcn_update_dpp(0, __builtin_bit_cast(int, v),
                                        0x160 | MASK, 0xF, 0xF, true);
    return fmaxf(v, __builtin_bit_cast(float, t));
}
template<int MASK>
static __device__ __forceinline__ float dpp_add(float v) {
    int t = __builtin_amdgcn_update_dpp(0, __builtin_bit_cast(int, v),
                                        0x160 | MASK, 0xF, 0xF, true);
    return v + __builtin_bit_cast(float, t);
}
static __device__ __forceinline__ float rowmax16(float v) {
    v = dpp_max<1>(v); v = dpp_max<2>(v); v = dpp_max<4>(v); return dpp_max<8>(v);
}
static __device__ __forceinline__ float rowsum16(float v) {
    v = dpp_add<1>(v); v = dpp_add<2>(v); v = dpp_add<4>(v); return dpp_add<8>(v);
}

// ---------------- Tensor Data Mover: 128-row x 32-elem bf16 tile -> padded LDS ----------------
// LDS layout: 40-element (80B) row stride == 64B data + 16B pad (pad_interval=64B, pad_amount=16B)
static __device__ __forceinline__ void tdm_load_tile(unsigned ldsAddr, const u16* g, int K) {
    unsigned long long ga = (unsigned long long)(uintptr_t)g;
    v4u g0;
    g0[0] = 1u;                                     // count=1, user descriptor
    g0[1] = ldsAddr;                                // LDS byte address
    g0[2] = (unsigned)ga;                           // global addr [31:0]
    g0[3] = (unsigned)((ga >> 32) & 0x1FFFFFFu) | (2u << 30);  // addr [56:32], type=2
    v8i g1;
    g1[0] = (int)((1u << 16)        // data_size = 2 bytes
                | (1u << 20)        // pad_enable
                | (3u << 22)        // pad_interval: 16 DWORDs = 64B
                | (3u << 25));      // pad_amount:   4 DWORDs = 16B
    g1[1] = (int)(((unsigned)K & 0xFFFFu) << 16);                  // tensor_dim0[15:0]
    g1[2] = (int)(((unsigned)K >> 16) | (128u << 16));             // tensor_dim0[31:16], tensor_dim1[15:0]=128
    g1[3] = (int)(32u << 16);                                      // tensor_dim1 hi=0, tile_dim0=32
    g1[4] = (int)128u;                                             // tile_dim1=128, tile_dim2=0
    g1[5] = (int)K;                                                // tensor_dim0_stride[31:0]
    g1[6] = 0;                                                     // stride hi, dim1_stride lo
    g1[7] = 0;
    v4i z = {0,0,0,0};
#if __clang_major__ >= 23
    v8i z8 = {0,0,0,0,0,0,0,0};
    __builtin_amdgcn_tensor_load_to_lds(g0, g1, z, z, z8, 0);
#else
    __builtin_amdgcn_tensor_load_to_lds(g0, g1, z, z, 0);
#endif
}

// ---------------- weight prep: fp32 (K x N) -> bf16 Bt (N x K) ----------------
__global__ __launch_bounds__(256) void wconv(const float* __restrict__ W, u16* __restrict__ Bt,
                                             int K, int N) {
    int idx = blockIdx.x * 256 + threadIdx.x;
    int n = idx / K, k = idx % K;
    Bt[idx] = f2bf(W[(size_t)k * N + n]);
}

__global__ __launch_bounds__(256) void copyf(float* __restrict__ dst, const float* __restrict__ src, int n) {
    int i = blockIdx.x * 256 + threadIdx.x;
    if (i < n) dst[i] = src[i];
}

// ---------------- LayerNorm fp32 -> bf16 (one block per row, D=1024) ----------------
__global__ __launch_bounds__(256) void ln_rows(const float* __restrict__ x, const float* __restrict__ g,
                                               const float* __restrict__ be, u16* __restrict__ out) {
    const int row = blockIdx.x, t = threadIdx.x;
    const float4* xr = (const float4*)(x + (size_t)row * 1024);
    float4 v = xr[t];
    float s = v.x + v.y + v.z + v.w;
    float q = v.x*v.x + v.y*v.y + v.z*v.z + v.w*v.w;
    __shared__ float s1[256], s2[256];
    s1[t] = s; s2[t] = q; __syncthreads();
    for (int o = 128; o > 0; o >>= 1) {
        if (t < o) { s1[t] += s1[t+o]; s2[t] += s2[t+o]; }
        __syncthreads();
    }
    float mean = s1[0] * (1.0f/1024.0f);
    float var  = s2[0] * (1.0f/1024.0f) - mean*mean;
    float rstd = rsqrtf(var + 1e-5f);
    int c = t*4;
    u16 o0 = f2bf((v.x-mean)*rstd*g[c+0] + be[c+0]);
    u16 o1 = f2bf((v.y-mean)*rstd*g[c+1] + be[c+1]);
    u16 o2 = f2bf((v.z-mean)*rstd*g[c+2] + be[c+2]);
    u16 o3 = f2bf((v.w-mean)*rstd*g[c+3] + be[c+3]);
    uint2 pk; pk.x = (unsigned)o0 | ((unsigned)o1<<16); pk.y = (unsigned)o2 | ((unsigned)o3<<16);
    *(uint2*)(out + (size_t)row*1024 + c) = pk;
}

// ---------------- fused: x2 = x1 + ocrow[b]; xn3 = LN(x2) ----------------
__global__ __launch_bounds__(256) void add_ln(const float* __restrict__ x1, const float* __restrict__ ocrow,
                                              const float* __restrict__ g, const float* __restrict__ be,
                                              float* __restrict__ x2, u16* __restrict__ xn) {
    const int row = blockIdx.x, t = threadIdx.x;
    const int b = row >> 11;  // S = 2048
    const float4* xr = (const float4*)(x1 + (size_t)row * 1024);
    const float4* orr = (const float4*)(ocrow + (size_t)b * 1024);
    float4 v = xr[t], o = orr[t];
    v.x += o.x; v.y += o.y; v.z += o.z; v.w += o.w;
    ((float4*)(x2 + (size_t)row*1024))[t] = v;
    float s = v.x + v.y + v.z + v.w;
    float q = v.x*v.x + v.y*v.y + v.z*v.z + v.w*v.w;
    __shared__ float s1[256], s2[256];
    s1[t] = s; s2[t] = q; __syncthreads();
    for (int off = 128; off > 0; off >>= 1) {
        if (t < off) { s1[t] += s1[t+off]; s2[t] += s2[t+off]; }
        __syncthreads();
    }
    float mean = s1[0] * (1.0f/1024.0f);
    float var  = s2[0] * (1.0f/1024.0f) - mean*mean;
    float rstd = rsqrtf(var + 1e-5f);
    int c = t*4;
    u16 o0 = f2bf((v.x-mean)*rstd*g[c+0] + be[c+0]);
    u16 o1 = f2bf((v.y-mean)*rstd*g[c+1] + be[c+1]);
    u16 o2 = f2bf((v.z-mean)*rstd*g[c+2] + be[c+2]);
    u16 o3 = f2bf((v.w-mean)*rstd*g[c+3] + be[c+3]);
    uint2 pk; pk.x = (unsigned)o0 | ((unsigned)o1<<16); pk.y = (unsigned)o2 | ((unsigned)o3<<16);
    *(uint2*)(xn + (size_t)row*1024 + c) = pk;
}

// ---------------- RoPE + head reshape ----------------
__global__ __launch_bounds__(256) void rope_reshape(const u16* __restrict__ qkv,
                                                    const float* __restrict__ cosp, const float* __restrict__ sinp,
                                                    u16* __restrict__ qb, u16* __restrict__ kb, u16* __restrict__ vt) {
    const int row = blockIdx.x, t = threadIdx.x;
    const int b = row >> 11, s = row & 2047;
    const u16* qr = qkv + (size_t)row * 3072;
#pragma unroll
    for (int i = 0; i < 2; ++i) {
        int p = t + 256*i;
        float c = cosp[(size_t)s*512 + p];
        float sn = sinp[(size_t)s*512 + p];
        float x1 = bf2f(qr[p]), x2 = bf2f(qr[p + 512]);
        float a = x1*c - x2*sn, bb = x1*sn + x2*c;
        int d1 = p, d2 = p + 512;
        qb[(((size_t)(b*16 + (d1>>6)))*2048 + s)*64 + (d1&63)] = f2bf(a);
        qb[(((size_t)(b*16 + (d2>>6)))*2048 + s)*64 + (d2&63)] = f2bf(bb);
        x1 = bf2f(qr[1024 + p]); x2 = bf2f(qr[1024 + p + 512]);
        a = x1*c - x2*sn; bb = x1*sn + x2*c;
        kb[(((size_t)(b*16 + (d1>>6)))*2048 + s)*64 + (d1&63)] = f2bf(a);
        kb[(((size_t)(b*16 + (d2>>6)))*2048 + s)*64 + (d2&63)] = f2bf(bb);
    }
#pragma unroll
    for (int i = 0; i < 4; ++i) {
        int d = t + 256*i;
        vt[(((size_t)(b*16 + (d>>6)))*64 + (d&63))*2048 + s] = qr[2048 + d];
    }
}

// ---------------- tiny row-vector GEMM ----------------
__global__ __launch_bounds__(256) void rowvec(const float* __restrict__ vin, const float* __restrict__ W,
                                              const float* __restrict__ bias, float* __restrict__ vout,
                                              int Dk, int N) {
    int n = blockIdx.x * 256 + threadIdx.x;
    const float* vr = vin + (size_t)blockIdx.y * Dk;
    float acc = bias[n];
    for (int k = 0; k < Dk; ++k) acc += vr[k] * W[(size_t)k * N + n];
    vout[(size_t)blockIdx.y * N + n] = acc;
}

// ---------------- bf16 WMMA GEMM with TDM double-buffered LDS staging --------
// MODE 0: bf16 out = acc+bias ; MODE 1: f32 out = acc+bias+res ; MODE 2: bf16 out = gelu(acc+bias)
template<int MODE>
__global__ __launch_bounds__(256) void gemm_bf16(const u16* __restrict__ A, const u16* __restrict__ Bt,
                                                 const float* __restrict__ bias, const float* __restrict__ res,
                                                 float* __restrict__ outF, u16* __restrict__ outB,
                                                 int M, int N, int K) {
    __shared__ u16 As[2][128*40];
    __shared__ u16 Bs[2][128*40];
    const int t = threadIdx.x;
    const int wid = t >> 5, lane = t & 31;
    const int wm = wid >> 2, wn = wid & 3;
    const int lm = lane & 15, half = lane >> 4;
    const int rowBase = blockIdx.y * 128;
    const int colBase = blockIdx.x * 128;
    const u16* aBase = A  + (size_t)rowBase * K;
    const u16* bBase = Bt + (size_t)colBase * K;

    v8f acc[4][2];
#pragma unroll
    for (int i = 0; i < 4; ++i)
#pragma unroll
        for (int j = 0; j < 2; ++j) acc[i][j] = v8zero();

    if (wid == 0) {
        tdm_load_tile((unsigned)(uintptr_t)&As[0][0], aBase, K);
        tdm_load_tile((unsigned)(uintptr_t)&Bs[0][0], bBase, K);
    }

    for (int kt = 0; kt < K; kt += 32) {
        const int cur = (kt >> 5) & 1;
        if (wid == 0) __builtin_amdgcn_s_wait_tensorcnt(0);
        __syncthreads();                       // current tiles visible to all waves
        if (wid == 0 && (kt + 32) < K) {       // prefetch next tiles via TDM (overlaps WMMA)
            tdm_load_tile((unsigned)(uintptr_t)&As[cur ^ 1][0], aBase + kt + 32, K);
            tdm_load_tile((unsigned)(uintptr_t)&Bs[cur ^ 1][0], bBase + kt + 32, K);
        }

        FragBF aF[4], bF[2];
#pragma unroll
        for (int i = 0; i < 4; ++i) {
            int r = (wm*64 + i*16 + lm) * 40;
            aF[i].q[0] = *(const uint4*)&As[cur][r + 8*half];
            aF[i].q[1] = *(const uint4*)&As[cur][r + 16 + 8*half];
        }
#pragma unroll
        for (int j = 0; j < 2; ++j) {
            int r = (wn*32 + j*16 + lm) * 40;
            bF[j].q[0] = *(const uint4*)&Bs[cur][r + 8*half];
            bF[j].q[1] = *(const uint4*)&Bs[cur][r + 16 + 8*half];
        }
#pragma unroll
        for (int i = 0; i < 4; ++i)
#pragma unroll
            for (int j = 0; j < 2; ++j)
                acc[i][j] = __builtin_amdgcn_wmma_f32_16x16x32_bf16(
                    false, aF[i].v, false, bF[j].v, (short)0, acc[i][j], false, false);
    }

#pragma unroll
    for (int i = 0; i < 4; ++i) {
#pragma unroll
        for (int j = 0; j < 2; ++j) {
            int row0 = rowBase + wm*64 + i*16 + 8*half;
            int col  = colBase + wn*32 + j*16 + lm;
            float bc = bias[col];
#pragma unroll
            for (int r = 0; r < 8; ++r) {
                float v = acc[i][j][r] + bc;
                size_t idx = (size_t)(row0 + r) * N + col;
                if (MODE == 0) outB[idx] = f2bf(v);
                else if (MODE == 1) outF[idx] = v + res[idx];
                else outB[idx] = f2bf(gelu_exact(v));
            }
        }
    }
}

// ---------------- flash attention, bf16 WMMA, online softmax (DPP reductions) ----------------
__global__ __launch_bounds__(128) void flash_attn(const u16* __restrict__ qb, const u16* __restrict__ kb,
                                                  const u16* __restrict__ vt, u16* __restrict__ o) {
    __shared__ u16 Pl[4*16*40];
    const int t = threadIdx.x;
    const int wid = t >> 5, lane = t & 31;
    const int lm = lane & 15, half = lane >> 4;
    const int bh = blockIdx.y;
    const int b = bh >> 4, h = bh & 15;
    const int qrow0 = blockIdx.x*64 + wid*16;

    FragBF qF[2];
#pragma unroll
    for (int f = 0; f < 2; ++f) {
        const u16* qp = qb + (((size_t)bh)*2048 + qrow0 + lm)*64 + 32*f;
        qF[f].q[0] = *(const uint4*)(qp + 8*half);
        qF[f].q[1] = *(const uint4*)(qp + 16 + 8*half);
    }

    v8f oacc[4];
#pragma unroll
    for (int s4 = 0; s4 < 4; ++s4) oacc[s4] = v8zero();
    float m8[8], l8[8];
#pragma unroll
    for (int r = 0; r < 8; ++r) { m8[r] = -1e30f; l8[r] = 0.0f; }

    const int pbase = wid * 640;

    for (int kt = 0; kt < 2048; kt += 32) {
        v8f sc[2]; sc[0] = v8zero(); sc[1] = v8zero();
#pragma unroll
        for (int sub = 0; sub < 2; ++sub) {
#pragma unroll
            for (int kh = 0; kh < 2; ++kh) {
                FragBF kF;
                const u16* kp = kb + (((size_t)bh)*2048 + kt + 16*sub + lm)*64 + 32*kh + 8*half;
                kF.q[0] = ((const uint4*)kp)[0];
                kF.q[1] = ((const uint4*)kp)[2];
                sc[sub] = __builtin_amdgcn_wmma_f32_16x16x32_bf16(
                    false, qF[kh].v, false, kF.v, (short)0, sc[sub], false, false);
            }
        }
#pragma unroll
        for (int r = 0; r < 8; ++r) {
            float a0 = sc[0][r] * 0.125f;
            float a1 = sc[1][r] * 0.125f;
            float mx = rowmax16(fmaxf(a0, a1));
            float mnew = fmaxf(m8[r], mx);
            float al = __expf(m8[r] - mnew);
            float p0 = __expf(a0 - mnew);
            float p1 = __expf(a1 - mnew);
            float rs = rowsum16(p0 + p1);
            l8[r] = l8[r]*al + rs;
            m8[r] = mnew;
#pragma unroll
            for (int s4 = 0; s4 < 4; ++s4) oacc[s4][r] *= al;
            Pl[pbase + (r + 8*half)*40 + lm]      = f2bf(p0);
            Pl[pbase + (r + 8*half)*40 + lm + 16] = f2bf(p1);
        }
        __syncthreads();
        FragBF pF;
        pF.q[0] = *(const uint4*)&Pl[pbase + lm*40 + 8*half];
        pF.q[1] = *(const uint4*)&Pl[pbase + lm*40 + 16 + 8*half];
#pragma unroll
        for (int s4 = 0; s4 < 4; ++s4) {
            FragBF vF;
            const u16* vp = vt + (((size_t)bh)*64 + 16*s4 + lm)*2048 + kt + 8*half;
            vF.q[0] = ((const uint4*)vp)[0];
            vF.q[1] = ((const uint4*)vp)[2];
            oacc[s4] = __builtin_amdgcn_wmma_f32_16x16x32_bf16(
                false, pF.v, false, vF.v, (short)0, oacc[s4], false, false);
        }
        __syncthreads();
    }
#pragma unroll
    for (int s4 = 0; s4 < 4; ++s4) {
#pragma unroll
        for (int r = 0; r < 8; ++r) {
            float val = oacc[s4][r] / l8[r];
            int srow = qrow0 + r + 8*half;
            o[((size_t)(b*2048 + srow))*1024 + h*64 + 16*s4 + lm] = f2bf(val);
        }
    }
}

// ---------------- host-side orchestration ----------------
extern "C" void kernel_launch(void* const* d_in, const int* in_sizes, int n_in,
                              void* d_out, int out_size, void* d_ws, size_t ws_size,
                              hipStream_t stream) {
    const float* x     = (const float*)d_in[0];
    const float* cond  = (const float*)d_in[1];
    const float* cosp  = (const float*)d_in[2];
    const float* sinp  = (const float*)d_in[3];
    const float* Wq = (const float*)d_in[4],  *bq = (const float*)d_in[5];
    const float* Wk = (const float*)d_in[6],  *bk = (const float*)d_in[7];
    const float* Wv = (const float*)d_in[8],  *bv = (const float*)d_in[9];
    const float* Wo = (const float*)d_in[10], *bo = (const float*)d_in[11];
    const float* Wvc = (const float*)d_in[16], *bvc = (const float*)d_in[17];
    const float* Woc = (const float*)d_in[18], *boc = (const float*)d_in[19];
    const float* W1 = (const float*)d_in[20], *b1 = (const float*)d_in[21];
    const float* W2 = (const float*)d_in[22], *b2 = (const float*)d_in[23];
    const float* g1 = (const float*)d_in[24], *be1 = (const float*)d_in[25];
    const float* g3 = (const float*)d_in[28], *be3 = (const float*)d_in[29];
    float* dout = (float*)d_out;
    char* ws = (char*)d_ws;

    u16*   btqkv  = (u16*)(ws + 0);                  //  6,291,456  (3072 x 1024)
    u16*   wot    = (u16*)(ws + 6291456);            //  2,097,152
    u16*   w1t    = (u16*)(ws + 8388608);            //  8,388,608
    u16*   w2t    = (u16*)(ws + 16777216);           //  8,388,608
    float* biasqkv= (float*)(ws + 25165824);         //  12,288
    float* vc     = (float*)(ws + 25178112);         //  8,192
    float* ocrow  = (float*)(ws + 25186304);         //  8,192
    u16*   xn1    = (u16*)(ws + 25194496);           //  8,388,608  (reused as xn3)
    u16*   qkvraw = (u16*)(ws + 33583104);           // 25,165,824  (reused: h spans qkvraw+qb)
    u16*   qbuf   = (u16*)(ws + 58748928);           //  8,388,608
    u16*   kbuf   = (u16*)(ws + 67137536);           //  8,388,608  (reused: x2 spans kb+vt)
    u16*   vtb    = (u16*)(ws + 75526144);           //  8,388,608
    u16*   obuf   = (u16*)(ws + 83914752);           //  8,388,608
    u16*   xn3    = xn1;
    u16*   hbuf   = qkvraw;
    float* x2     = (float*)kbuf;

    wconv<<<4096, 256, 0, stream>>>(Wq, btqkv,             1024, 1024);
    wconv<<<4096, 256, 0, stream>>>(Wk, btqkv + 1024*1024, 1024, 1024);
    wconv<<<4096, 256, 0, stream>>>(Wv, btqkv + 2048*1024, 1024, 1024);
    wconv<<<4096, 256, 0, stream>>>(Wo, wot,               1024, 1024);
    wconv<<<16384,256, 0, stream>>>(W1, w1t,               1024, 4096);
    wconv<<<16384,256, 0, stream>>>(W2, w2t,               4096, 1024);
    copyf<<<4, 256, 0, stream>>>(biasqkv,        bq, 1024);
    copyf<<<4, 256, 0, stream>>>(biasqkv + 1024, bk, 1024);
    copyf<<<4, 256, 0, stream>>>(biasqkv + 2048, bv, 1024);

    ln_rows<<<4096, 256, 0, stream>>>(x, g1, be1, xn1);

    gemm_bf16<0><<<dim3(24, 32), 256, 0, stream>>>(xn1, btqkv, biasqkv, nullptr,
                                                   nullptr, qkvraw, 4096, 3072, 1024);
    rope_reshape<<<4096, 256, 0, stream>>>(qkvraw, cosp, sinp, qbuf, kbuf, vtb);

    flash_attn<<<dim3(32, 32), 128, 0, stream>>>(qbuf, kbuf, vtb, obuf);

    gemm_bf16<1><<<dim3(8, 32), 256, 0, stream>>>(obuf, wot, bo, x,
                                                  dout, nullptr, 4096, 1024, 1024);

    rowvec<<<dim3(4, 2), 256, 0, stream>>>(cond, Wvc, bvc, vc,    1024, 1024);
    rowvec<<<dim3(4, 2), 256, 0, stream>>>(vc,   Woc, boc, ocrow, 1024, 1024);

    add_ln<<<4096, 256, 0, stream>>>(dout, ocrow, g3, be3, x2, xn3);

    gemm_bf16<2><<<dim3(32, 32), 256, 0, stream>>>(xn3, w1t, b1, nullptr,
                                                   nullptr, hbuf, 4096, 4096, 1024);
    gemm_bf16<1><<<dim3(8, 32), 256, 0, stream>>>(hbuf, w2t, b2, x2,
                                                  dout, nullptr, 4096, 1024, 4096);
}
